// _RFCN_79328045957594
// MI455X (gfx1250) — matrix operand order repaired and checked by reading, not compiled, not run on hardware
//
#include <hip/hip_runtime.h>
#include <hip/hip_bf16.h>

// ---------------------------------------------------------------------------
// Types for CDNA5 WMMA (wave32, gfx1250)
// ---------------------------------------------------------------------------
typedef __attribute__((ext_vector_type(8)))  float          v8f;
typedef __attribute__((ext_vector_type(16))) __bf16         v16bf;
typedef __attribute__((ext_vector_type(8)))  unsigned short ushort8;

union Frag {
    v16bf   bf;
    ushort8 h[2];
};

union Pack8 {
    ushort8      v;
    unsigned int u[4];
};

// Set to 0 if the gfx1250 assembler rejects v_cvt_pk_bf16_f32.
#define USE_ASM_CVT_PK_BF16 1

__device__ __forceinline__ unsigned short f2bf(float x) {
    // round-to-nearest-even fp32 -> bf16 (scalar)
    unsigned int u = __float_as_uint(x);
    u += 0x7FFFu + ((u >> 16) & 1u);
    return (unsigned short)(u >> 16);
}

__device__ __forceinline__ unsigned int pk2bf(float a, float b) {
    // packed fp32x2 -> bf16x2 (lo = a, hi = b)
#if __has_builtin(__builtin_amdgcn_cvt_pk_bf16_f32)
    auto p = __builtin_amdgcn_cvt_pk_bf16_f32(a, b);
    unsigned int u;
    __builtin_memcpy(&u, &p, sizeof(u));
    return u;
#elif USE_ASM_CVT_PK_BF16
    unsigned int u;
    asm("v_cvt_pk_bf16_f32 %0, %1, %2" : "=v"(u) : "v"(a), "v"(b));
    return u;
#else
    unsigned int ua = __float_as_uint(a);
    ua += 0x7FFFu + ((ua >> 16) & 1u);
    unsigned int ub = __float_as_uint(b);
    ub += 0x7FFFu + ((ub >> 16) & 1u);
    // result = {ub[31:16], ua[31:16]}
    return __builtin_amdgcn_perm(ub, ua, 0x07060302u);
#endif
}

// ---------------------------------------------------------------------------
// Batched GEMM:  C[b] = (relu?) A * B[b]
//   A : [M x K] fp32 row-major (weights, shared across batch)
//   B : [K x N] per batch, TB = float (convert) or unsigned short (bf16 pass)
//   C : [M x N] per batch, TC = unsigned short (bf16 out) or float
// Workgroup: 256 thr = 8 waves; tile 128(M) x 128(N); K-step 32.
// Waves in 4(M) x 2(N) grid; each wave: 2x4 v_wmma_f32_16x16x32_bf16 (32x64).
// Software-pipelined: tile t+1 global loads issued before tile t's barrier.
// ---------------------------------------------------------------------------
template <typename TB, typename TC, bool RELU>
__global__ __launch_bounds__(256) void gemm_wmma_bf16(
    const float* __restrict__ A,
    const TB*    __restrict__ B,
    TC*          __restrict__ C,
    int M, int K, int N,
    long long strideB, long long strideC)
{
    const int tid    = threadIdx.x;
    const int blockN = blockIdx.x * 128;
    const int blockM = blockIdx.y * 128;
    const TB* Bb = B + (long long)blockIdx.z * strideB;
    TC*       Cb = C + (long long)blockIdx.z * strideC;

    // 40-half row stride (80 B): 16B-aligned rows, bank-spread, b128-friendly
    __shared__ alignas(16) unsigned short As[128 * 40];   // [m][k]
    __shared__ alignas(16) unsigned short Bs[128 * 40];   // [n][k] (transposed)

    const int lane = tid & 31;
    const int wv   = tid >> 5;
    const int wm   = wv & 3;           // wave M slot: 0..3 -> +32*wm
    const int wn   = wv >> 2;          // wave N slot: 0..1 -> +64*wn
    const int lrow = lane & 15;
    const int hi   = lane >> 4;        // lane half (0/1)
    const int ka0  = hi << 3;          // A frag: 8-half chunks at ka0, ka0+16
    const int kb0  = hi << 3;          // B frag: 8-half chunks at kb0, kb0+16

    // A staging footprint: 1 row x 16 cols; B: 2 cols (n2,n2+1) x 8 rows
    const int arow = tid >> 1;
    const int acol = (tid & 1) << 4;
    const int grow = blockM + arow;
    const bool aval = grow < M;
    const float* aBase = A + (long long)grow * K + acol;

    const int n2 = (tid & 63) << 1;    // 0..126
    const int kq = (tid >> 6) << 3;    // 0,8,16,24

    // Partial M-tile: zero-fill A LDS once; in-loop path never zero-fills.
    if (blockM + 128 > M) {
        const ushort8 z = {0, 0, 0, 0, 0, 0, 0, 0};
        for (int idx = tid; idx < 128 * 5; idx += 256)
            ((ushort8*)As)[idx] = z;
        __syncthreads();
    }

    // -------- software-pipeline registers --------
    float4 ar[4];
    union { float2 f[8]; unsigned int u[8]; } br;

    auto loadA = [&](int kb) {
        const float* ap = aBase + kb;
        ar[0] = *(const float4*)(ap + 0);
        ar[1] = *(const float4*)(ap + 4);
        ar[2] = *(const float4*)(ap + 8);
        ar[3] = *(const float4*)(ap + 12);
        __builtin_prefetch(ap + 32, 0, 1);             // speculative, fault-safe
    };
    auto loadB = [&](int kb) {
        if constexpr (sizeof(TB) == 4) {
            const float* bp = (const float*)Bb + (long long)(kb + kq) * N + blockN + n2;
            #pragma unroll
            for (int r = 0; r < 8; ++r)
                br.f[r] = *(const float2*)(bp + (long long)r * N);
            __builtin_prefetch(bp + 32LL * N, 0, 1);
        } else {
            const unsigned short* bp =
                (const unsigned short*)Bb + (long long)(kb + kq) * N + blockN + n2;
            #pragma unroll
            for (int r = 0; r < 8; ++r)
                br.u[r] = *(const unsigned int*)(bp + (long long)r * N);
            __builtin_prefetch(bp + 32LL * N, 0, 1);
        }
    };

    if (aval) loadA(0);
    loadB(0);

    v8f acc[2][4] = {};

    for (int kb = 0; kb < K; kb += 32) {
        // ---- commit staged registers to LDS (fp32 -> bf16 where needed) ----
        if (aval) {
            Pack8 p0, p1;
            p0.u[0] = pk2bf(ar[0].x, ar[0].y); p0.u[1] = pk2bf(ar[0].z, ar[0].w);
            p0.u[2] = pk2bf(ar[1].x, ar[1].y); p0.u[3] = pk2bf(ar[1].z, ar[1].w);
            p1.u[0] = pk2bf(ar[2].x, ar[2].y); p1.u[1] = pk2bf(ar[2].z, ar[2].w);
            p1.u[2] = pk2bf(ar[3].x, ar[3].y); p1.u[3] = pk2bf(ar[3].z, ar[3].w);
            *(ushort8*)&As[arow * 40 + acol]     = p0.v;
            *(ushort8*)&As[arow * 40 + acol + 8] = p1.v;
        }
        {
            Pack8 c0, c1;   // c0: column n2, c1: column n2+1; 8 k-halves each
            if constexpr (sizeof(TB) == 4) {
                #pragma unroll
                for (int q = 0; q < 4; ++q) {
                    c0.u[q] = pk2bf(br.f[2 * q].x, br.f[2 * q + 1].x);
                    c1.u[q] = pk2bf(br.f[2 * q].y, br.f[2 * q + 1].y);
                }
            } else {
                #pragma unroll
                for (int q = 0; q < 4; ++q) {
                    // lo halves of (ld0,ld1) / hi halves of (ld0,ld1)
                    c0.u[q] = __builtin_amdgcn_perm(br.u[2 * q + 1], br.u[2 * q], 0x05040100u);
                    c1.u[q] = __builtin_amdgcn_perm(br.u[2 * q + 1], br.u[2 * q], 0x07060302u);
                }
            }
            *(ushort8*)&Bs[(n2 + 0) * 40 + kq] = c0.v;
            *(ushort8*)&Bs[(n2 + 1) * 40 + kq] = c1.v;
        }

        // ---- issue next tile's global loads (overlap with barrier + WMMA) ----
        if (kb + 32 < K) {
            if (aval) loadA(kb + 32);
            loadB(kb + 32);
        }
        __syncthreads();

        // ---- fragments per ISA 16-bit layouts ----
        Frag a[2], b[4];
        #pragma unroll
        for (int i = 0; i < 2; ++i) {
            const int r = (wm * 32 + i * 16 + lrow) * 40;
            a[i].h[0] = *(const ushort8*)&As[r + ka0];        // K 0-7  / 8-15
            a[i].h[1] = *(const ushort8*)&As[r + ka0 + 16];   // K 16-23/ 24-31
        }
        #pragma unroll
        for (int j = 0; j < 4; ++j) {
            const int c = (wn * 64 + j * 16 + lrow) * 40;
            b[j].h[0] = *(const ushort8*)&Bs[c + kb0];        // K 0-7  / 8-15
            b[j].h[1] = *(const ushort8*)&Bs[c + kb0 + 16];   // K 16-23/ 24-31
        }

        #pragma unroll
        for (int i = 0; i < 2; ++i)
            #pragma unroll
            for (int j = 0; j < 4; ++j)
                acc[i][j] = __builtin_amdgcn_wmma_f32_16x16x32_bf16(
                    false, a[i].bf, false, b[j].bf, (short)0, acc[i][j], false, false);

        __syncthreads();
    }

    // ---- store C: reg r, lane L -> (M = r + 8*(L/16), N = L%16) ----
    #pragma unroll
    for (int i = 0; i < 2; ++i) {
        #pragma unroll
        for (int j = 0; j < 4; ++j) {
            const int colg = blockN + wn * 64 + j * 16 + lrow;
            #pragma unroll
            for (int r = 0; r < 8; ++r) {
                const int rowg = blockM + wm * 32 + i * 16 + r + (hi << 3);
                if (rowg < M) {
                    float v = acc[i][j][r];
                    if (RELU) v = fmaxf(v, 0.0f);
                    if constexpr (sizeof(TC) == 2)
                        Cb[(long long)rowg * N + colg] = f2bf(v);
                    else
                        Cb[(long long)rowg * N + colg] = v;
                }
            }
        }
    }
}

// ---------------------------------------------------------------------------
// PSRoI average pool + mean over 7x7 bins.
// grid = (nRois, outDim); block = 64 threads; threads 0..48 -> one bin each.
// Channel index == d*49 + ph*7 + pw == d*49 + tid (reference reshape order).
// Deterministic LDS tree reduction (no float atomics).
// ---------------------------------------------------------------------------
__global__ __launch_bounds__(64) void psroi_reduce(
    const float* __restrict__ feat,   // [4, C, 64, 64]
    const float* __restrict__ rois,   // [nRois, 5]
    float* __restrict__ dst,          // [nRois, outDim]
    int C, int outDim)
{
    const int n   = blockIdx.x;
    const int d   = blockIdx.y;
    const int tid = threadIdx.x;

    float val = 0.0f;
    if (tid < 49) {
        const float* r = rois + (size_t)n * 5;
        const int   b  = (int)r[0];
        const float x1 = floorf(r[1] + 0.5f) * 0.0625f;
        const float y1 = floorf(r[2] + 0.5f) * 0.0625f;
        const float x2 = floorf(r[3] + 1.5f) * 0.0625f;   // floor((v+1)+0.5)
        const float y2 = floorf(r[4] + 1.5f) * 0.0625f;
        const float bw = fmaxf(x2 - x1, 0.1f) * (1.0f / 7.0f);
        const float bh = fmaxf(y2 - y1, 0.1f) * (1.0f / 7.0f);

        const int ph = tid / 7;
        const int pw = tid % 7;
        const int hs  = (int)fminf(fmaxf(floorf((float)ph * bh + y1), 0.0f), 64.0f);
        const int he  = (int)fminf(fmaxf(ceilf((float)(ph + 1) * bh + y1), 0.0f), 64.0f);
        const int wsx = (int)fminf(fmaxf(floorf((float)pw * bw + x1), 0.0f), 64.0f);
        const int wex = (int)fminf(fmaxf(ceilf((float)(pw + 1) * bw + x1), 0.0f), 64.0f);

        const int c = d * 49 + tid;
        const float* f = feat + ((size_t)b * C + c) * 4096;
        float s = 0.0f;
        for (int h = hs; h < he; ++h)
            for (int w = wsx; w < wex; ++w)
                s += f[h * 64 + w];
        const int area = (he - hs) * (wex - wsx);
        val = (area > 0) ? s / (float)area : 0.0f;
    }

    __shared__ float red[64];
    red[tid] = val;
    __syncthreads();
    #pragma unroll
    for (int s = 32; s > 0; s >>= 1) {
        if (tid < s) red[tid] += red[tid + s];
        __syncthreads();
    }
    if (tid == 0) dst[(size_t)n * outDim + d] = red[0] * (1.0f / 49.0f);
}

// ---------------------------------------------------------------------------
// Row softmax over 21 classes (one thread per RoI; trivial work).
// ---------------------------------------------------------------------------
__global__ void softmax_rows(const float* __restrict__ sc,
                             float* __restrict__ out, int nR)
{
    const int n = blockIdx.x * blockDim.x + threadIdx.x;
    if (n >= nR) return;
    const float* s = sc + (size_t)n * 21;
    float m = s[0];
    #pragma unroll
    for (int d = 1; d < 21; ++d) m = fmaxf(m, s[d]);
    float e[21];
    float sum = 0.0f;
    #pragma unroll
    for (int d = 0; d < 21; ++d) { e[d] = expf(s[d] - m); sum += e[d]; }
    const float inv = 1.0f / sum;
    #pragma unroll
    for (int d = 0; d < 21; ++d) out[(size_t)n * 21 + d] = e[d] * inv;
}

// ---------------------------------------------------------------------------
// Host launcher
// ---------------------------------------------------------------------------
extern "C" void kernel_launch(void* const* d_in, const int* in_sizes, int n_in,
                              void* d_out, int out_size, void* d_ws, size_t ws_size,
                              hipStream_t stream)
{
    (void)n_in; (void)out_size; (void)ws_size;

    const float* base_feat = (const float*)d_in[0];  // [4, 2048, 64, 64]
    const float* rois      = (const float*)d_in[1];  // [1200, 5]
    const float* w_conv1   = (const float*)d_in[2];  // [1024, 2048]
    const float* w_cls     = (const float*)d_in[3];  // [1029, 1024]
    const float* w_bbox    = (const float*)d_in[4];  // [4116, 1024]
    float* out = (float*)d_out;

    const int nR   = in_sizes[1] / 5;       // 1200
    const int Bn   = 4;
    const int HW   = 64 * 64;               // 4096
    const int C0   = 2048;
    const int C1   = 1024;
    const int Ccls = 21 * 49;               // 1029
    const int Cbb  = 84 * 49;               // 4116

    // Workspace layout
    char* ws = (char*)d_ws;
    unsigned short* xbf = (unsigned short*)ws;                 // bf16 x: 4*1024*4096
    size_t off = (size_t)Bn * C1 * HW * sizeof(unsigned short);
    float* clsf = (float*)(ws + off); off += (size_t)Bn * Ccls * HW * sizeof(float);
    float* bbf  = (float*)(ws + off); off += (size_t)Bn * Cbb  * HW * sizeof(float);
    float* scores = (float*)(ws + off);                        // nR * 21

    const dim3 blk(256);

    // GEMM1: x = relu(w_conv1 * base_feat)   [fp32 in -> bf16 out]
    gemm_wmma_bf16<float, unsigned short, true>
        <<<dim3(HW / 128, (C1 + 127) / 128, Bn), blk, 0, stream>>>(
            w_conv1, base_feat, xbf, C1, C0, HW,
            (long long)C0 * HW, (long long)C1 * HW);

    // GEMM2: cls_feat = w_cls * x            [bf16 in -> fp32 out]
    gemm_wmma_bf16<unsigned short, float, false>
        <<<dim3(HW / 128, (Ccls + 127) / 128, Bn), blk, 0, stream>>>(
            w_cls, xbf, clsf, Ccls, C1, HW,
            (long long)C1 * HW, (long long)Ccls * HW);

    // GEMM3: bbox_feat = w_bbox * x          [bf16 in -> fp32 out]
    gemm_wmma_bf16<unsigned short, float, false>
        <<<dim3(HW / 128, (Cbb + 127) / 128, Bn), blk, 0, stream>>>(
            w_bbox, xbf, bbf, Cbb, C1, HW,
            (long long)C1 * HW, (long long)Cbb * HW);

    // PSRoI pool cls -> scores, then softmax -> out[0 .. nR*21)
    psroi_reduce<<<dim3(nR, 21), 64, 0, stream>>>(clsf, rois, scores, Ccls, 21);
    softmax_rows<<<(nR + 255) / 256, 256, 0, stream>>>(scores, out, nR);

    // PSRoI pool bbox -> out[nR*21 .. nR*21 + nR*84)
    psroi_reduce<<<dim3(nR, 84), 64, 0, stream>>>(bbf, rois, out + (size_t)nR * 21, Cbb, 84);
}
